// AAGNN_A_25615184954167
// MI455X (gfx1250) — matrix operand order repaired and checked by reading, not compiled, not run on hardware
//
#include <hip/hip_runtime.h>
#include <hip/hip_bf16.h>

// ---------------------------------------------------------------------------
// AAGNN forward, MI455X (gfx1250).
//
// Algebraic reduction: A is 0/1 with self loops, so
//   att(i,j) = e_i / S_i  (edges j != i),  att(i,i) = d_i / S_i
//   e_i = exp(lrelu(zi_i)), d_i = exp(lrelu(zi_i + zj_i)),
//   S_i = e_i*(deg_i - 1) + d_i
//   (att@z)_i = (e_i*(A_i.Z - z_i) + d_i*z_i) / S_i
// Dominant cost: one 256MB pass over A producing deg = A.1 and S = A@Z.
// S = A@Z via v_wmma_f32_16x16x32_bf16 with Z split bf16 hi+lo (A is exactly
// representable in bf16), keeping the kernel HBM-bound (~11us @ 23.3TB/s).
// M-tile = 32 rows per wave so each B fragment (L2-resident) feeds two A
// fragments, halving L2 B-traffic (0.5GB) vs the 16-row version.
// ---------------------------------------------------------------------------

#define NN 8192
#define IN_F 128
#define OUT_F 64
#define NEG_SLOPE 0.01f

typedef __bf16 v16bf __attribute__((ext_vector_type(16)));
typedef float  v8f   __attribute__((ext_vector_type(8)));
typedef float  v4f   __attribute__((ext_vector_type(4)));

__device__ __forceinline__ float lrelu(float x) { return x > 0.f ? x : NEG_SLOPE * x; }

// B-fragment packing for V_WMMA_F32_16X16X32_BF16 (B is 32(K) x 16(N)):
//   lane = (n & 15) + 16*((k >> 3) & 1), slot-in-lane = (k & 7) + 8*((k >> 4) & 1), chunk = k >> 5.
__device__ __forceinline__ size_t bpack_idx(int k, int n) {
  int lane = (n & 15) | (((k >> 3) & 1) << 4);
  return ((size_t)(((k >> 5) * 4 + (n >> 4)) * 32 + lane)) * 16 + ((k & 7) | (((k >> 4) & 1) << 3));
}

// ---- k0: zero the S accumulator (d_out) and deg ---------------------------
__global__ void k0_zero(float* __restrict__ S, float* __restrict__ DEG) {
  int idx = blockIdx.x * blockDim.x + threadIdx.x;
  if (idx < NN * OUT_F) S[idx] = 0.f;
  if (idx < NN) DEG[idx] = 0.f;
}

// ---- k1: Z = X @ W^T + b, emit f32 Z and packed bf16 hi/lo fragments ------
__global__ void k1_linear(const float* __restrict__ X, const float* __restrict__ W,
                          const float* __restrict__ b, float* __restrict__ Z,
                          __bf16* __restrict__ Zhi, __bf16* __restrict__ Zlo) {
  int gid = blockIdx.x * blockDim.x + threadIdx.x;   // NN*OUT_F threads
  int i = gid >> 6, o = gid & 63;
  const v4f* xr = (const v4f*)(X + (size_t)i * IN_F);
  const v4f* wr = (const v4f*)(W + (size_t)o * IN_F);
  float acc = 0.f;
#pragma unroll
  for (int k = 0; k < IN_F / 4; ++k) {
    v4f x = xr[k], w = wr[k];
    acc = fmaf(x[0], w[0], acc);
    acc = fmaf(x[1], w[1], acc);
    acc = fmaf(x[2], w[2], acc);
    acc = fmaf(x[3], w[3], acc);
  }
  float z = acc + b[o];
  Z[gid] = z;
  __bf16 zh = (__bf16)z;
  __bf16 zl = (__bf16)(z - (float)zh);
  size_t p = bpack_idx(i, o);      // K dim of A@Z = node index, N dim = feature
  Zhi[p] = zh;
  Zlo[p] = zl;
}

// ---- k2: per-row scalars e_i, d_i (one wave32 per row) --------------------
__global__ void k2_rowscalars(const float* __restrict__ Z, const float* __restrict__ a1,
                              const float* __restrict__ a2, float* __restrict__ E,
                              float* __restrict__ D) {
  int row = (blockIdx.x * blockDim.x + threadIdx.x) >> 5;
  int lane = threadIdx.x & 31;
  const float* zr = Z + (size_t)row * OUT_F;
  float z0 = zr[lane], z1 = zr[lane + 32];
  float p1 = fmaf(a1[lane], z0, a1[lane + 32] * z1);
  float p2 = fmaf(a2[lane], z0, a2[lane + 32] * z1);
#pragma unroll
  for (int off = 16; off > 0; off >>= 1) {
    p1 += __shfl_xor(p1, off, 32);
    p2 += __shfl_xor(p2, off, 32);
  }
  if (lane == 0) {
    E[row] = expf(lrelu(p1));
    D[row] = expf(lrelu(p1 + p2));
  }
}

// Load one 16x32 bf16 A fragment (rows base..base+15) and accumulate its f32
// row-degree partial. A-fragment layout (16-bit A, 16x32): lane(m,h):
// VGPR v<4 -> K = 8h+2v(+1); v>=4 -> K = 16+8h+2(v-4)(+1).
// Relative to (arow = row + 8h): slots 0..7 <- k 0..7, slots 8..15 <- k 16..23.
__device__ __forceinline__ v16bf load_afrag(const float* __restrict__ arow, int kb,
                                            float& degp) {
  v4f x0 = __builtin_nontemporal_load((const v4f*)(arow + kb));
  v4f x1 = __builtin_nontemporal_load((const v4f*)(arow + kb + 4));
  v4f x2 = __builtin_nontemporal_load((const v4f*)(arow + kb + 16));
  v4f x3 = __builtin_nontemporal_load((const v4f*)(arow + kb + 20));
  v16bf a;
#pragma unroll
  for (int j = 0; j < 4; ++j) {
    degp += x0[j] + x1[j] + x2[j] + x3[j];
    a[j]      = (__bf16)x0[j];
    a[4 + j]  = (__bf16)x1[j];
    a[8 + j]  = (__bf16)x2[j];
    a[12 + j] = (__bf16)x3[j];
  }
  return a;
}

#define WMMA_BF16(C, Afrag, Bfrag) \
  C = __builtin_amdgcn_wmma_f32_16x16x32_bf16(false, Afrag, false, Bfrag, (short)0, C, false, false)

// ---- k3: S += A@Z (bf16 hi/lo WMMA), deg += A.1 ---------------------------
// One wave per (32-row M-tile, 512-wide K-slice): grid (256, 16).
__global__ void __launch_bounds__(32) k3_spmm(const float* __restrict__ A,
                                              const __bf16* __restrict__ Zhi,
                                              const __bf16* __restrict__ Zlo,
                                              float* __restrict__ S,
                                              float* __restrict__ DEG) {
  const int lane = threadIdx.x & 31;
  const int m = lane & 15, hh = lane >> 4;
  const int r0 = blockIdx.x * 32;
  const int kb0 = blockIdx.y * 512;

  v8f c0 = {}, c1 = {}, c2 = {}, c3 = {};   // rows r0..r0+15, nt 0..3
  v8f c4 = {}, c5 = {}, c6 = {}, c7 = {};   // rows r0+16..r0+31, nt 0..3
  float degp0 = 0.f, degp1 = 0.f;

  const float* arow0 = A + (size_t)(r0 + m) * NN + hh * 8;
  const float* arow1 = arow0 + (size_t)16 * NN;
  const v16bf* bhBase = (const v16bf*)Zhi;
  const v16bf* blBase = (const v16bf*)Zlo;

  for (int ck = 0; ck < 16; ++ck) {
    const int kb = kb0 + ck * 32;
    v16bf a0 = load_afrag(arow0, kb, degp0);
    v16bf a1 = load_afrag(arow1, kb, degp1);
    const size_t base = (size_t)(kb >> 5) * 128 + lane;  // (chunk*4 + nt)*32 + lane

    v16bf bh0 = bhBase[base + 0],  bl0 = blBase[base + 0];
    WMMA_BF16(c0, a0, bh0); WMMA_BF16(c4, a1, bh0);
    WMMA_BF16(c0, a0, bl0); WMMA_BF16(c4, a1, bl0);

    v16bf bh1 = bhBase[base + 32], bl1 = blBase[base + 32];
    WMMA_BF16(c1, a0, bh1); WMMA_BF16(c5, a1, bh1);
    WMMA_BF16(c1, a0, bl1); WMMA_BF16(c5, a1, bl1);

    v16bf bh2 = bhBase[base + 64], bl2 = blBase[base + 64];
    WMMA_BF16(c2, a0, bh2); WMMA_BF16(c6, a1, bh2);
    WMMA_BF16(c2, a0, bl2); WMMA_BF16(c6, a1, bl2);

    v16bf bh3 = bhBase[base + 96], bl3 = blBase[base + 96];
    WMMA_BF16(c3, a0, bh3); WMMA_BF16(c7, a1, bh3);
    WMMA_BF16(c3, a0, bl3); WMMA_BF16(c7, a1, bl3);
  }

  // deg: lanes (m,0) and (m,1) hold complementary K halves
  degp0 += __shfl_xor(degp0, 16, 32);
  degp1 += __shfl_xor(degp1, 16, 32);
  if (hh == 0) {
    atomicAdd(&DEG[r0 + m], degp0);
    atomicAdd(&DEG[r0 + 16 + m], degp1);
  }

  // C/D layout: VGPR v, lanes 0-15 -> M = v, lanes 16-31 -> M = v + 8; N = lane&15
  const int col = lane & 15;
#pragma unroll
  for (int v = 0; v < 8; ++v) {
    const size_t row0 = (size_t)(r0 + v + 8 * hh);
    const size_t row1 = row0 + 16;
    atomicAdd(&S[row0 * OUT_F + col +  0], c0[v]);
    atomicAdd(&S[row0 * OUT_F + col + 16], c1[v]);
    atomicAdd(&S[row0 * OUT_F + col + 32], c2[v]);
    atomicAdd(&S[row0 * OUT_F + col + 48], c3[v]);
    atomicAdd(&S[row1 * OUT_F + col +  0], c4[v]);
    atomicAdd(&S[row1 * OUT_F + col + 16], c5[v]);
    atomicAdd(&S[row1 * OUT_F + col + 32], c6[v]);
    atomicAdd(&S[row1 * OUT_F + col + 48], c7[v]);
  }
}

// ---- k4: h = relu(z - att@z), in place on d_out ---------------------------
__global__ void k4_final(float* __restrict__ Sout, const float* __restrict__ Z,
                         const float* __restrict__ E, const float* __restrict__ D,
                         const float* __restrict__ DEG) {
  int idx = blockIdx.x * blockDim.x + threadIdx.x;
  int i = idx >> 6;
  float s = Sout[idx], z = Z[idx];
  float e = E[i], dd = D[i], dg = DEG[i];
  float denom = e * (dg - 1.0f) + dd;
  float attz = (e * (s - z) + dd * z) / denom;
  float hv = z - attz;
  Sout[idx] = hv > 0.f ? hv : 0.f;
}

extern "C" void kernel_launch(void* const* d_in, const int* in_sizes, int n_in,
                              void* d_out, int out_size, void* d_ws, size_t ws_size,
                              hipStream_t stream) {
  const float* X  = (const float*)d_in[0];
  const float* A  = (const float*)d_in[1];
  const float* W  = (const float*)d_in[2];
  const float* b  = (const float*)d_in[3];
  const float* a1 = (const float*)d_in[4];
  const float* a2 = (const float*)d_in[5];
  float* out = (float*)d_out;

  char* ws = (char*)d_ws;
  float*  Z   = (float*)(ws);                                     // 2 MB
  __bf16* Zhi = (__bf16*)(ws + (size_t)NN * OUT_F * 4);           // 1 MB
  __bf16* Zlo = (__bf16*)(ws + (size_t)NN * OUT_F * 6);           // 1 MB
  float*  E   = (float*)(ws + (size_t)NN * OUT_F * 8);            // 32 KB
  float*  D   = E + NN;                                           // 32 KB
  float*  DEG = D + NN;                                           // 32 KB

  k0_zero<<<(NN * OUT_F + 255) / 256, 256, 0, stream>>>(out, DEG);
  k1_linear<<<(NN * OUT_F + 255) / 256, 256, 0, stream>>>(X, W, b, Z, Zhi, Zlo);
  k2_rowscalars<<<NN * 32 / 256, 256, 0, stream>>>(Z, a1, a2, E, D);
  dim3 g3(NN / 32, 16);  // 256 M-tiles (32 rows) x 16 K-splits, 1 wave each
  k3_spmm<<<g3, 32, 0, stream>>>(A, Zhi, Zlo, out, DEG);
  k4_final<<<(NN * OUT_F + 255) / 256, 256, 0, stream>>>(out, Z, E, D, DEG);
}
